// Nnue_43525198578243
// MI455X (gfx1250) — matrix operand order repaired and checked by reading, not compiled, not run on hardware
//
#include <hip/hip_runtime.h>
#include <hip/hip_bf16.h>
#include <math.h>

typedef __attribute__((ext_vector_type(16))) _Float16 v16h;
typedef __attribute__((ext_vector_type(8)))  _Float16 v8h;
typedef __attribute__((ext_vector_type(8)))  float    v8f;
typedef __attribute__((ext_vector_type(4)))  float    v4f;

#define F_IN    45056
#define BATCH   2048
#define M_TOT   (BATCH * 2)      // 4096 rows (2048 batch x 2 perspectives)
#define N1      256
#define KT      64               // K-tile per barrier (2 WMMA K-steps)
#define KSPLIT  2                // K-split factor -> grid = 128 * KSPLIT
#define KCHUNK  (F_IN / KSPLIT)  // 22528 (multiple of KT)

// ---------------------------------------------------------------------------
// Kernel 0: W1 f32 -> f16 (one-time prep into workspace).
// ---------------------------------------------------------------------------
__global__ __launch_bounds__(256) void nnue_w1_to_f16(
    const float* __restrict__ w1, _Float16* __restrict__ w1h) {
    const int i = (blockIdx.x * 256 + threadIdx.x) * 8;
    v4f lo = *(const v4f*)(w1 + i);
    v4f hi = *(const v4f*)(w1 + i + 4);
    v8h o;
    o[0] = (_Float16)lo[0]; o[1] = (_Float16)lo[1];
    o[2] = (_Float16)lo[2]; o[3] = (_Float16)lo[3];
    o[4] = (_Float16)hi[0]; o[5] = (_Float16)hi[1];
    o[6] = (_Float16)hi[2]; o[7] = (_Float16)hi[3];
    *(v8h*)(w1h + i) = o;
}

// ---------------------------------------------------------------------------
// Kernel 1: layer-1 GEMM partial sums.
//   part[ks][4096,256] += X[4096, kchunk] @ W1h^T[kchunk, 256]
// Block: 256 threads (8 waves); tile 32(M) x 256(N) x KCHUNK(K).
// grid.x = 128 M-tiles * KSPLIT.  X read from HBM exactly once.
// Staging pipeline: load tile k+2 -> regs | cvt+ds_store tile k+1 | wmma tile k
// ---------------------------------------------------------------------------
__device__ __forceinline__ void store_cvt(_Float16* __restrict__ dst,
                                          int sr, int sc, v4f a, v4f b) {
    v8h h;
    h[0] = (_Float16)a[0]; h[1] = (_Float16)a[1];
    h[2] = (_Float16)a[2]; h[3] = (_Float16)a[3];
    h[4] = (_Float16)b[0]; h[5] = (_Float16)b[1];
    h[6] = (_Float16)b[2]; h[7] = (_Float16)b[3];
    *(v8h*)(dst + sr * KT + sc) = h;    // ds_store_b128
}

__global__ __launch_bounds__(256) void nnue_layer1(
    const float* __restrict__ X, const _Float16* __restrict__ W1h,
    float* __restrict__ part) {
    __shared__ _Float16 smemA[2][32 * KT];

    const int tid  = threadIdx.x;
    const int lane = tid & 31;
    const int wave = tid >> 5;
    const int mt   = blockIdx.x % 128;
    const int ks   = blockIdx.x / 128;
    const int m0   = mt * 32;
    const int kbeg = ks * KCHUNK;
    const int kend = kbeg + KCHUNK;
    const int wn0  = wave * 32;

    // staging: thread loads 8 consecutive f32 of the 32xKT X tile
    const int sr = tid >> 3;
    const int sc = (tid & 7) * 8;
    const float* xrow = X + (size_t)(m0 + sr) * F_IN + sc;

    // fragment coords (16-bit WMMA layouts)
    const int mrow = lane & 15;        // M (A) / N (B) within 16-subtile
    const int ka0  = (lane >> 4) * 16; // A: K base 0 or 16
    const int kb0  = (lane >> 4) * 16; // B: K base 0 or 16

    v8f acc[2][2] = {};                // [m_sub][n_sub]

    // ---- pipeline prologue: tile0 -> LDS[0], tile1 loads in flight ----
    v4f rb0{}, rb1{};
    {
        v4f a = *(const v4f*)(xrow + kbeg);
        v4f b = *(const v4f*)(xrow + kbeg + 4);
        store_cvt(smemA[0], sr, sc, a, b);
        rb0 = *(const v4f*)(xrow + kbeg + KT);
        rb1 = *(const v4f*)(xrow + kbeg + KT + 4);
    }
    __syncthreads();

    int buf = 0;
    for (int kb = kbeg; kb < kend; kb += KT) {
        // issue loads for tile k+2 (waited on only next iteration)
        v4f rn0{}, rn1{};
        const bool have_nn = (kb + 2 * KT) < kend;
        if (have_nn) {
            rn0 = *(const v4f*)(xrow + kb + 2 * KT);
            rn1 = *(const v4f*)(xrow + kb + 2 * KT + 4);
        }
        // convert + ds_store tile k+1 (its loads were issued last iter)
        if (kb + KT < kend)
            store_cvt(smemA[buf ^ 1], sr, sc, rb0, rb1);

        // ---- compute tile k: 2 K-steps of 32, 4 WMMAs each ----
#pragma unroll
        for (int k2 = 0; k2 < KT; k2 += 32) {
            v16h bf[2];
#pragma unroll
            for (int t = 0; t < 2; ++t) {
                const _Float16* bp =
                    W1h + (size_t)(wn0 + t * 16 + mrow) * F_IN + kb + k2 + kb0;
                bf[t] = *(const v16h*)bp;          // 2x global_load_b128
                __builtin_prefetch(bp + KT, 0, 1); // next iter's B fragment
            }
            v16h af[2];
#pragma unroll
            for (int s = 0; s < 2; ++s) {
                const _Float16* ap =
                    &smemA[buf][(s * 16 + mrow) * KT + k2 + ka0];
                v16h a;
                *(v8h*)&a       = *(const v8h*)ap;        // K0..K0+7
                *((v8h*)&a + 1) = *(const v8h*)(ap + 8);  // K0+8..K0+15
                af[s] = a;
            }
#pragma unroll
            for (int s = 0; s < 2; ++s)
#pragma unroll
                for (int t = 0; t < 2; ++t)
                    acc[s][t] = __builtin_amdgcn_wmma_f32_16x16x32_f16(
                        false, af[s], false, bf[t], (short)0, acc[s][t],
                        false, false);
        }

        __syncthreads();
        if (have_nn) { rb0 = rn0; rb1 = rn1; }
        buf ^= 1;
    }

    // ---- epilogue: raw f32 partial sums (bias/relu folded into tail) ----
    float* pout = part + (size_t)ks * ((size_t)M_TOT * N1);
    const int rbase = m0 + (lane >> 4) * 8;
#pragma unroll
    for (int s = 0; s < 2; ++s) {
#pragma unroll
        for (int t = 0; t < 2; ++t) {
            const int col = wn0 + t * 16 + mrow;
            float* outp = pout + (size_t)(rbase + s * 16) * N1 + col;
#pragma unroll
            for (int e = 0; e < 8; ++e)
                outp[(size_t)e * N1] = acc[s][t][e];
        }
    }
}

// ---------------------------------------------------------------------------
// Kernel 2: combine K-split partials (+b1, relu) then layers 2-4 + sigmoid.
// One wave per batch row; the 512-wide activation row is staged in LDS.
// ---------------------------------------------------------------------------
__global__ __launch_bounds__(256) void nnue_tail(
    const float* __restrict__ part, const float* __restrict__ b1,
    const float* __restrict__ W2,   const float* __restrict__ b2,
    const float* __restrict__ W3,   const float* __restrict__ b3,
    const float* __restrict__ W4,   const float* __restrict__ b4,
    float* __restrict__ out) {
    __shared__ float arow[8][512];
    const int lane = threadIdx.x & 31;
    const int wv   = threadIdx.x >> 5;
    const int row  = blockIdx.x * 8 + wv;

    // a[row, k] = relu(part0 + part1 + b1[k % 256])  (flat layouts coincide)
    const float* p0 = part + (size_t)row * 512;
    const float* p1 = p0 + (size_t)M_TOT * N1;
#pragma unroll 4
    for (int j = lane; j < 512; j += 32) {
        float av = p0[j] + p1[j] + b1[j & 255];
        arow[wv][j] = av > 0.f ? av : 0.f;
    }
    __syncthreads();

    const float* w2r = W2 + (size_t)lane * 512;
    float acc = b2[lane];
#pragma unroll 8
    for (int k = 0; k < 512; ++k) acc = fmaf(w2r[k], arow[wv][k], acc);
    float h2 = acc > 0.f ? acc : 0.f;

    const float* w3r = W3 + lane * 32;
    float acc3 = b3[lane];
#pragma unroll
    for (int k = 0; k < 32; ++k) acc3 = fmaf(w3r[k], __shfl(h2, k, 32), acc3);
    float h3 = acc3 > 0.f ? acc3 : 0.f;

    float p = W4[lane] * h3;
#pragma unroll
    for (int off = 16; off; off >>= 1) p += __shfl_xor(p, off, 32);
    if (lane == 0) out[row] = 1.f / (1.f + __expf(-(p + b4[0])));
}

// ---------------------------------------------------------------------------
extern "C" void kernel_launch(void* const* d_in, const int* in_sizes, int n_in,
                              void* d_out, int out_size, void* d_ws, size_t ws_size,
                              hipStream_t stream) {
    const float* x  = (const float*)d_in[0];
    const float* W1 = (const float*)d_in[1];
    const float* b1 = (const float*)d_in[2];
    const float* W2 = (const float*)d_in[3];
    const float* b2 = (const float*)d_in[4];
    const float* W3 = (const float*)d_in[5];
    const float* b3 = (const float*)d_in[6];
    const float* W4 = (const float*)d_in[7];
    const float* b4 = (const float*)d_in[8];

    _Float16* w1h  = (_Float16*)d_ws;                   // 23,068,672 B
    float*    part = (float*)((char*)d_ws + 23068672);  // KSPLIT * 4 MiB

    nnue_w1_to_f16<<<(N1 * F_IN) / (256 * 8), 256, 0, stream>>>(W1, w1h);
    nnue_layer1  <<<128 * KSPLIT,             256, 0, stream>>>(x, w1h, part);
    nnue_tail    <<<BATCH / 8,                256, 0, stream>>>(part, b1, W2, b2,
                                                               W3, b3, W4, b4,
                                                               (float*)d_out);
}
